// BaseModel_57045755625809
// MI455X (gfx1250) — matrix operand order repaired and checked by reading, not compiled, not run on hardware
//
#include <hip/hip_runtime.h>
#include <hip/hip_bf16.h>

// ---------------------------------------------------------------------------
// Types for CDNA5 WMMA
// ---------------------------------------------------------------------------
typedef __attribute__((ext_vector_type(8)))  __bf16 v8bf;
typedef __attribute__((ext_vector_type(16))) __bf16 v16bf;
typedef __attribute__((ext_vector_type(8)))  float  v8f;
typedef __attribute__((ext_vector_type(4)))  int    v4i;

#define GRU_B 2048
#define GRU_T 13
#define GRU_D 188
#define GRU_H 256
#define GRU_L 8
#define GRU_BT (GRU_B * GRU_T)   // 26624
#define GRU_G  (3 * GRU_H)       // 768
#define KPAD   192               // 188 padded to multiple of 32

#define BM 128
#define BN 128
#define BK 32

// gfx1250 async global->LDS path (ASYNCcnt), with sync fallback
#if defined(__gfx1250__) && __has_builtin(__builtin_amdgcn_global_load_async_to_lds_b128) && __has_builtin(__builtin_amdgcn_s_wait_asynccnt)
#define USE_ASYNC_LDS 1
#else
#define USE_ASYNC_LDS 0
#endif

#define ASG __attribute__((address_space(1)))
#define ASL __attribute__((address_space(3)))

__device__ __forceinline__ unsigned short f2bf(float f) {
    unsigned u = __builtin_bit_cast(unsigned, f);
    unsigned r = (u + 0x7FFFu + ((u >> 16) & 1u)) >> 16;   // RNE
    return (unsigned short)r;
}

__device__ __forceinline__ float sigm(float x) { return 1.0f / (1.0f + expf(-x)); }

// ---------------------------------------------------------------------------
// f32 -> bf16 cast with optional row zero-padding (Ls -> Ld columns)
// ---------------------------------------------------------------------------
__global__ void cast_pad_bf16_kernel(const float* __restrict__ src,
                                     unsigned short* __restrict__ dst,
                                     int rows, int Ls, int Ld) {
    int i = blockIdx.x * blockDim.x + threadIdx.x;
    if (i >= rows * Ld) return;
    int r = i / Ld, c = i - r * Ld;
    float v = (c < Ls) ? src[r * Ls + c] : 0.0f;
    dst[i] = f2bf(v);
}

__global__ void zero_f32_kernel(float* __restrict__ p, int n) {
    int i = blockIdx.x * blockDim.x + threadIdx.x;
    if (i < n) p[i] = 0.0f;
}

__global__ void zero_h_kernel(float* __restrict__ hf, unsigned short* __restrict__ hbf, int n) {
    int i = blockIdx.x * blockDim.x + threadIdx.x;
    if (i < n) { hf[i] = 0.0f; hbf[i] = 0; }
}

// ---------------------------------------------------------------------------
// Stage one 128x32 A tile + 128x32 W tile into LDS.
// 256 threads, each moves 2 chunks of 16B per tile (async on gfx1250).
// ---------------------------------------------------------------------------
__device__ __forceinline__ void stage_tiles(const unsigned short* __restrict__ A, int lda,
                                            const unsigned short* __restrict__ W, int ldw,
                                            int bm, int bn, int k0, int tid,
                                            unsigned short* As, unsigned short* Ws) {
    #pragma unroll
    for (int i = 0; i < 2; ++i) {
        int c  = tid + 256 * i;       // 0..511
        int r  = c >> 2;              // 0..127
        int cc = (c & 3) * 8;         // 0,8,16,24
        const unsigned short* ga = &A[(size_t)(bm + r) * lda + k0 + cc];
        const unsigned short* gw = &W[(size_t)(bn + r) * ldw + k0 + cc];
#if USE_ASYNC_LDS
        __builtin_amdgcn_global_load_async_to_lds_b128(
            (ASG v4i*)ga, (ASL v4i*)&As[r * BK + cc], 0, 0);
        __builtin_amdgcn_global_load_async_to_lds_b128(
            (ASG v4i*)gw, (ASL v4i*)&Ws[r * BK + cc], 0, 0);
#else
        *(uint4*)&As[r * BK + cc] = *(const uint4*)ga;
        *(uint4*)&Ws[r * BK + cc] = *(const uint4*)gw;
#endif
    }
}

#if USE_ASYNC_LDS
#define ASYNC_WAIT() __builtin_amdgcn_s_wait_asynccnt(0)
#else
#define ASYNC_WAIT()
#endif

// ---------------------------------------------------------------------------
// WMMA GEMM: C[M,N] = A[M,K](bf16, row-major) * W[N,K](bf16, row-major)^T + bias[N]
// Block tile 128x128, BK=32, double-buffered LDS staged via async-to-LDS.
// 8 waves in a 4x2 grid; each wave owns a 32x64 tile => 2x4 grid of
// v_wmma_f32_16x16x32_bf16 per K-step. All dims must be tile-multiples.
// ---------------------------------------------------------------------------
__global__ void __launch_bounds__(256)
gemm_bf16_wmma_kernel(const unsigned short* __restrict__ A, int lda,
                      const unsigned short* __restrict__ W, int ldw,
                      const float* __restrict__ bias,
                      float* __restrict__ C, int ldc,
                      int K) {
    __shared__ __align__(16) unsigned short As[2][BM * BK];   // 2 x 8KB
    __shared__ __align__(16) unsigned short Ws[2][BN * BK];   // 2 x 8KB

    const int tid  = threadIdx.x;
    const int wave = tid >> 5;      // 0..7
    const int lane = tid & 31;
    const int wm   = wave >> 1;     // 0..3  -> M offset wm*32
    const int wn   = wave & 1;      // 0..1  -> N offset wn*64
    const int bm   = blockIdx.x * BM;
    const int bn   = blockIdx.y * BN;

    v8f acc[2][4];
    #pragma unroll
    for (int i = 0; i < 2; ++i)
        #pragma unroll
        for (int j = 0; j < 4; ++j)
            acc[i][j] = (v8f){0.f, 0.f, 0.f, 0.f, 0.f, 0.f, 0.f, 0.f};

    // A fragment lane mapping (ISA 7.12.2, 16-bit A 16x32):
    //   lane<16: elems 0..7 = K kh..kh+7, elems 8..15 = K kh+16..kh+23, kh=0
    //   lane>=16: kh=8
    const int ar = lane & 15;
    const int kh = (lane >> 4) * 8;
    // B fragment lane mapping (K x N, 32x16):
    //   lane<16: N=lane, K=0..7,16..23 ; lane>=16: N=lane-16, K=8..15,24..31
    const int bnl = lane & 15;
    const int kb  = (lane >> 4) * 8;

    const int ntiles = K / BK;

    // prologue: stage tile 0 into buffer 0
    stage_tiles(A, lda, W, ldw, bm, bn, 0, tid, As[0], Ws[0]);

    for (int it = 0; it < ntiles; ++it) {
        const int cur = it & 1;
        ASYNC_WAIT();          // own async ops for tile 'it' landed in LDS
        __syncthreads();       // all waves' tile data visible; prev reads done

        if (it + 1 < ntiles)   // overlap: stage next tile into other buffer
            stage_tiles(A, lda, W, ldw, bm, bn, (it + 1) * BK, tid,
                        As[cur ^ 1], Ws[cur ^ 1]);

        // ---- build fragments from current buffer ----
        v16bf afrag[2], bfrag[4];
        #pragma unroll
        for (int i = 0; i < 2; ++i) {
            const unsigned short* rowp = &As[cur][(wm * 32 + i * 16 + ar) * BK];
            v8bf lo = *reinterpret_cast<const v8bf*>(&rowp[kh]);
            v8bf hi = *reinterpret_cast<const v8bf*>(&rowp[kh + 16]);
            afrag[i] = __builtin_shufflevector(lo, hi, 0, 1, 2, 3, 4, 5, 6, 7,
                                               8, 9, 10, 11, 12, 13, 14, 15);
        }
        #pragma unroll
        for (int j = 0; j < 4; ++j) {
            const unsigned short* rowp = &Ws[cur][(wn * 64 + j * 16 + bnl) * BK];
            v8bf lo = *reinterpret_cast<const v8bf*>(&rowp[kb]);
            v8bf hi = *reinterpret_cast<const v8bf*>(&rowp[kb + 16]);
            bfrag[j] = __builtin_shufflevector(lo, hi, 0, 1, 2, 3, 4, 5, 6, 7,
                                               8, 9, 10, 11, 12, 13, 14, 15);
        }

        // ---- 2x4 WMMA ----
        #pragma unroll
        for (int i = 0; i < 2; ++i)
            #pragma unroll
            for (int j = 0; j < 4; ++j)
                acc[i][j] = __builtin_amdgcn_wmma_f32_16x16x32_bf16(
                    false, afrag[i], false, bfrag[j], (short)0, acc[i][j],
                    false, false);
    }

    // ---- epilogue: C layout (ISA 7.12.2 32-bit C/D 16x16):
    //   lane<16: N=lane, VGPR v -> M=v ; lane>=16: N=lane-16, VGPR v -> M=8+v
    const int mofs = (lane >> 4) * 8;
    #pragma unroll
    for (int i = 0; i < 2; ++i) {
        #pragma unroll
        for (int j = 0; j < 4; ++j) {
            int cn = bn + wn * 64 + j * 16 + (lane & 15);
            float bv = bias[cn];
            int mbase = bm + wm * 32 + i * 16 + mofs;
            #pragma unroll
            for (int v = 0; v < 8; ++v)
                C[(size_t)(mbase + v) * ldc + cn] = acc[i][j][v] + bv;
        }
    }
}

// ---------------------------------------------------------------------------
// GRU gate math for one time step (PyTorch r,z,n ordering).
// One thread per (b, n), n in [0,256).
// ---------------------------------------------------------------------------
__global__ void gru_gate_kernel(const float* __restrict__ gx,   // (B*T, 768)
                                const float* __restrict__ gh,   // (B, 768)
                                float* __restrict__ h_f,        // (B,256) in/out
                                unsigned short* __restrict__ h_bf,
                                unsigned short* __restrict__ out_bf, // (B*T, 512)
                                float* __restrict__ hsum,       // (B,256)
                                int t, int outOffset, int lastFlag) {
    int idx = blockIdx.x * blockDim.x + threadIdx.x;       // b*256 + n
    int b = idx >> 8, n = idx & 255;
    size_t gxr = (size_t)(b * GRU_T + t) * GRU_G;
    size_t ghr = (size_t)b * GRU_G;
    float r  = sigm(gx[gxr + n]           + gh[ghr + n]);
    float z  = sigm(gx[gxr + GRU_H + n]   + gh[ghr + GRU_H + n]);
    float nn = tanhf(gx[gxr + 2 * GRU_H + n] + r * gh[ghr + 2 * GRU_H + n]);
    float h  = h_f[idx];
    float hn = (1.0f - z) * nn + z * h;
    h_f[idx]  = hn;
    h_bf[idx] = f2bf(hn);
    out_bf[(size_t)(b * GRU_T + t) * (2 * GRU_H) + outOffset + n] = f2bf(hn);
    if (lastFlag) hsum[idx] += hn * (1.0f / 16.0f);
}

// ---------------------------------------------------------------------------
// MLP head: one 64-thread block per batch row.
// ---------------------------------------------------------------------------
__global__ void head_kernel(const float* __restrict__ hsum,
                            const float* __restrict__ W1, const float* __restrict__ b1,
                            const float* __restrict__ W2, const float* __restrict__ b2,
                            const float* __restrict__ Wc, const float* __restrict__ bc,
                            float* __restrict__ out) {
    __shared__ float sh[256];
    __shared__ float a1[64];
    __shared__ float a2[32];
    int b = blockIdx.x, tid = threadIdx.x;
    for (int i = tid; i < 256; i += 64) sh[i] = hsum[(size_t)b * 256 + i];
    __syncthreads();
    float acc = b1[tid];
    for (int k = 0; k < 256; ++k) acc += W1[tid * 256 + k] * sh[k];
    a1[tid] = acc * sigm(acc);
    __syncthreads();
    if (tid < 32) {
        float acc2 = b2[tid];
        for (int k = 0; k < 64; ++k) acc2 += W2[tid * 64 + k] * a1[k];
        a2[tid] = acc2 * sigm(acc2);
    }
    __syncthreads();
    if (tid == 0) {
        float acc3 = bc[0];
        for (int k = 0; k < 32; ++k) acc3 += Wc[k] * a2[k];
        out[b] = acc3;
    }
}

// ---------------------------------------------------------------------------
// Host-side orchestration
// ---------------------------------------------------------------------------
extern "C" void kernel_launch(void* const* d_in, const int* in_sizes, int n_in,
                              void* d_out, int out_size, void* d_ws, size_t ws_size,
                              hipStream_t stream) {
    (void)in_sizes; (void)n_in; (void)out_size; (void)ws_size;

    const float* x         = (const float*)d_in[0];   // (B,T,188)
    const float* W_ih0     = (const float*)d_in[1];   // (2,768,188)
    const float* W_ih_rest = (const float*)d_in[2];   // (7,2,768,512)
    const float* W_hh      = (const float*)d_in[3];   // (8,2,768,256)
    const float* b_ih      = (const float*)d_in[4];   // (8,2,768)
    const float* b_hh      = (const float*)d_in[5];   // (8,2,768)
    const float* W1 = (const float*)d_in[6];
    const float* b1 = (const float*)d_in[7];
    const float* W2 = (const float*)d_in[8];
    const float* b2 = (const float*)d_in[9];
    const float* Wc = (const float*)d_in[10];
    const float* bc = (const float*)d_in[11];
    float* out = (float*)d_out;

    // workspace carve-up (256B aligned)
    char* ws = (char*)d_ws;
    size_t off = 0;
    auto alloc = [&](size_t bytes) -> char* {
        char* p = ws + off;
        off += (bytes + 255) & ~(size_t)255;
        return p;
    };
    unsigned short* xpad   = (unsigned short*)alloc((size_t)GRU_BT * KPAD * 2);
    unsigned short* wih0b  = (unsigned short*)alloc((size_t)2 * GRU_G * KPAD * 2);
    unsigned short* wrestb = (unsigned short*)alloc((size_t)7 * 2 * GRU_G * 512 * 2);
    unsigned short* whhb   = (unsigned short*)alloc((size_t)8 * 2 * GRU_G * 256 * 2);
    unsigned short* inpA   = (unsigned short*)alloc((size_t)GRU_BT * 512 * 2);
    unsigned short* inpB   = (unsigned short*)alloc((size_t)GRU_BT * 512 * 2);
    float*          gx     = (float*)alloc((size_t)GRU_BT * GRU_G * 4);
    float*          gh     = (float*)alloc((size_t)GRU_B * GRU_G * 4);
    float*          h_f    = (float*)alloc((size_t)GRU_B * GRU_H * 4);
    unsigned short* h_bf   = (unsigned short*)alloc((size_t)GRU_B * GRU_H * 2);
    float*          hsum   = (float*)alloc((size_t)GRU_B * GRU_H * 4);

    const int BH = GRU_B * GRU_H;  // 524288

    // --- casts + init ---
    {
        int n;
        n = GRU_BT * KPAD;
        cast_pad_bf16_kernel<<<(n + 255) / 256, 256, 0, stream>>>(x, xpad, GRU_BT, GRU_D, KPAD);
        n = 2 * GRU_G * KPAD;
        cast_pad_bf16_kernel<<<(n + 255) / 256, 256, 0, stream>>>(W_ih0, wih0b, 2 * GRU_G, GRU_D, KPAD);
        n = 7 * 2 * GRU_G * 512;
        cast_pad_bf16_kernel<<<(n + 255) / 256, 256, 0, stream>>>(W_ih_rest, wrestb, 7 * 2 * GRU_G, 512, 512);
        n = 8 * 2 * GRU_G * 256;
        cast_pad_bf16_kernel<<<(n + 255) / 256, 256, 0, stream>>>(W_hh, whhb, 8 * 2 * GRU_G, 256, 256);
        zero_f32_kernel<<<(BH + 255) / 256, 256, 0, stream>>>(hsum, BH);
    }

    const unsigned short* Acur = xpad;
    int ldaCur = KPAD, Kcur = KPAD;

    for (int l = 0; l < GRU_L; ++l) {
        unsigned short* inp_next = (l & 1) ? inpB : inpA;
        for (int dir = 0; dir < 2; ++dir) {
            const unsigned short* Wi = (l == 0)
                ? wih0b + (size_t)dir * GRU_G * KPAD
                : wrestb + ((size_t)(l - 1) * 2 + dir) * GRU_G * 512;
            const float* bi = b_ih + ((size_t)l * 2 + dir) * GRU_G;
            const float* bh = b_hh + ((size_t)l * 2 + dir) * GRU_G;
            const unsigned short* Wh = whhb + ((size_t)l * 2 + dir) * GRU_G * 256;

            // gx = inp @ W_ih^T + b_ih   (M=26624, N=768, K=Kcur)
            {
                dim3 grid(GRU_BT / BM, GRU_G / BN);
                gemm_bf16_wmma_kernel<<<grid, 256, 0, stream>>>(
                    Acur, ldaCur, Wi, Kcur, bi, gx, GRU_G, Kcur);
            }

            // h0 = 0
            zero_h_kernel<<<(BH + 255) / 256, 256, 0, stream>>>(h_f, h_bf, BH);

            // sequential scan
            for (int s = 0; s < GRU_T; ++s) {
                int t = dir ? (GRU_T - 1 - s) : s;
                // gh = h @ W_hh^T + b_hh  (M=2048, N=768, K=256)
                dim3 grid(GRU_B / BM, GRU_G / BN);
                gemm_bf16_wmma_kernel<<<grid, 256, 0, stream>>>(
                    h_bf, GRU_H, Wh, GRU_H, bh, gh, GRU_G, GRU_H);
                gru_gate_kernel<<<BH / 256, 256, 0, stream>>>(
                    gx, gh, h_f, h_bf, inp_next, hsum, t, dir * GRU_H,
                    (s == GRU_T - 1) ? 1 : 0);
            }
        }
        Acur = inp_next;
        ldaCur = 2 * GRU_H;
        Kcur = 2 * GRU_H;
    }

    head_kernel<<<GRU_B, 64, 0, stream>>>(hsum, W1, b1, W2, b2, Wc, bc, out);
}
